// TemporalAttention_14611478741531
// MI455X (gfx1250) — compile-verified, matrix operand order checked by reading, not run on hardware
//
#include <hip/hip_runtime.h>

typedef __attribute__((ext_vector_type(16))) _Float16 v16h;
typedef __attribute__((ext_vector_type(8)))  _Float16 h8;
typedef __attribute__((ext_vector_type(8)))  float    v8f;

#define D_MODEL 1024
#define N_HEADS 16
#define HEAD_DIM 64
#define BATCH 2
#define SEQ 2048
#define ROWS (BATCH * SEQ)

// ---------------------------------------------------------------------------
// 16-bit WMMA A/B fragment in AMD table K order.
// Element e (0..15): K = (e<8 ? 8*hf+e : 16+8*hf+(e-8)).  Two 16B loads.
// ---------------------------------------------------------------------------
__device__ __forceinline__ v16h load_frag_k32(const _Float16* __restrict__ rowk0, int hf) {
    h8 lo = *reinterpret_cast<const h8*>(rowk0 + hf * 8);
    h8 hi = *reinterpret_cast<const h8*>(rowk0 + 16 + hf * 8);
    return __builtin_shufflevector(lo, hi, 0, 1, 2, 3, 4, 5, 6, 7,
                                           8, 9, 10, 11, 12, 13, 14, 15);
}

// ---------------------------------------------------------------------------
// Prep: x16 = f16(x); xt16 = f16(x + time_gaps*Wt + bt)
// ---------------------------------------------------------------------------
__global__ __launch_bounds__(256) void prep_kernel(
    const float* __restrict__ x, const float* __restrict__ tg,
    const float* __restrict__ Wt, const float* __restrict__ bt,
    _Float16* __restrict__ x16, _Float16* __restrict__ xt16, int rows) {
    int tid = blockIdx.x * blockDim.x + threadIdx.x;
    int row = tid / (D_MODEL / 4);
    int d0 = (tid % (D_MODEL / 4)) * 4;
    if (row >= rows) return;
    float g = tg[row];
    const float* xr = x + (size_t)row * D_MODEL;
#pragma unroll
    for (int j = 0; j < 4; ++j) {
        int d = d0 + j;
        float xv = xr[d];
        x16[(size_t)row * D_MODEL + d]  = (_Float16)xv;
        xt16[(size_t)row * D_MODEL + d] = (_Float16)(xv + g * Wt[d] + bt[d]);
    }
}

__global__ __launch_bounds__(256) void f32_to_f16_kernel(
    const float* __restrict__ src, _Float16* __restrict__ dst, int n) {
    int i = blockIdx.x * blockDim.x + threadIdx.x;
    if (i < n) dst[i] = (_Float16)src[i];
}

// ---------------------------------------------------------------------------
// GEMM: C[M,N] = cscale * (A[M,K] * B[N,K]^T + bias[N])
// Block: 256 threads (8 waves); tile 256(M) x 64(N); wave = 32(M) x 64(N)
// (8 WMMA per k-step).  B tile DOUBLE-BUFFERED in LDS via
// global_load_async_to_lds_b128 (ASYNCcnt): the async fetch of tile k+1
// overlaps the 8 WMMAs on tile k; one s_wait_asynccnt + one barrier per step.
// MODE 0: f32 row-major.  MODE 1: f16 row-major.
// MODE 2: f16 transposed per head: Vt[((b*H+h)*64+d)*SEQ + s] (16B stores).
// ---------------------------------------------------------------------------
template <int MODE>
__global__ __launch_bounds__(256) void gemm_bias_wmma(
    const _Float16* __restrict__ A, const _Float16* __restrict__ Bw,
    const float* __restrict__ bias, void* __restrict__ Cout,
    float cscale, int M, int N, int K) {
    __shared__ _Float16 sB[2][64][40];       // ping-pong, 32 k + 16B pad/row

    const int lane = threadIdx.x & 31;
    const int wave = threadIdx.x >> 5;
    const int hf   = lane >> 4;
    const int ln   = lane & 15;
    const int tilesN = N >> 6;
    const int bm = blockIdx.x / tilesN;
    const int bn = blockIdx.x % tilesN;
    const int m0 = bm * 256 + wave * 32;
    const int n0 = bn * 64;

    v8f acc[2][4] = {};
    const _Float16* arow0 = A + (size_t)(m0 + ln) * K;
    const _Float16* arow1 = A + (size_t)(m0 + 16 + ln) * K;

    // cooperative B-tile loader: 64 rows x 32 k, 16B per thread, async to LDS
    const int brow = threadIdx.x >> 2;       // 0..63
    const int bcol = (threadIdx.x & 3) << 3; // 0,8,16,24
    const _Float16* bsrc = Bw + (size_t)(n0 + brow) * K + bcol;
    unsigned la[2] = {(unsigned)(uintptr_t)&sB[0][brow][bcol],
                      (unsigned)(uintptr_t)&sB[1][brow][bcol]};

    // prologue: fetch tile 0
    {
        unsigned long long ga = (unsigned long long)(uintptr_t)bsrc;
        asm volatile("global_load_async_to_lds_b128 %0, %1, off"
                     :: "v"(la[0]), "v"(ga) : "memory");
    }

    int ib = 0;
    for (int k0 = 0; k0 < K; k0 += 32, ib ^= 1) {
        asm volatile("s_wait_asynccnt 0" ::: "memory");  // tile k0 in LDS
        __syncthreads();          // visible to all; prev-iter reads complete

        if (k0 + 32 < K) {        // kick off next tile, overlapped w/ compute
            unsigned long long ga =
                (unsigned long long)(uintptr_t)(bsrc + k0 + 32);
            asm volatile("global_load_async_to_lds_b128 %0, %1, off"
                         :: "v"(la[ib ^ 1]), "v"(ga) : "memory");
        }
        if (k0 + 64 <= K) {
            __builtin_prefetch(arow0 + k0 + 64, 0, 0);
            __builtin_prefetch(arow1 + k0 + 64, 0, 0);
        }

        v16h af0 = load_frag_k32(arow0 + k0, hf);
        v16h af1 = load_frag_k32(arow1 + k0, hf);
#pragma unroll
        for (int t = 0; t < 4; ++t) {
            v16h bf = load_frag_k32(&sB[ib][t * 16 + ln][0], hf);
            acc[0][t] = __builtin_amdgcn_wmma_f32_16x16x32_f16(
                false, af0, false, bf, (short)0, acc[0][t], false, false);
            acc[1][t] = __builtin_amdgcn_wmma_f32_16x16x32_f16(
                false, af1, false, bf, (short)0, acc[1][t], false, false);
        }
    }

#pragma unroll
    for (int mt = 0; mt < 2; ++mt) {
        const int mbase = m0 + mt * 16;
        if (MODE == 2) {
#pragma unroll
            for (int t = 0; t < 4; ++t) {
                int n = n0 + t * 16 + ln;
                int h = n >> 6, d = n & (HEAD_DIM - 1);
                float bv = bias[n];
                int m = mbase + 8 * hf;       // first of 8 consecutive rows (s)
                int b = m >> 11, s = m & (SEQ - 1);
                h8 out;
#pragma unroll
                for (int r = 0; r < 8; ++r)
                    out[r] = (_Float16)((acc[mt][t][r] + bv) * cscale);
                _Float16* dst = ((_Float16*)Cout) +
                    ((size_t)((b * N_HEADS + h) * HEAD_DIM + d)) * SEQ + s;
                *reinterpret_cast<h8*>(dst) = out;
            }
        } else {
#pragma unroll
            for (int t = 0; t < 4; ++t) {
                int n = n0 + t * 16 + ln;
                float bv = bias[n];
#pragma unroll
                for (int r = 0; r < 8; ++r) {
                    int m = mbase + r + 8 * hf;
                    float val = (acc[mt][t][r] + bv) * cscale;
                    if (MODE == 1) ((_Float16*)Cout)[(size_t)m * N + n] = (_Float16)val;
                    else           ((float*)Cout)[(size_t)m * N + n]    = val;
                }
            }
        }
    }
}

// ---------------------------------------------------------------------------
// Flash attention (causal + padding mask), transposed-score formulation.
// One wave per 16-query tile; 8 waves (128 queries of one (b,h)) per block.
// All 8 fragments of a kv chunk are issued before the first WMMA so the
// waits are incremental.  Mask logic runs only on the single diagonal chunk
// (wave-uniform branch); Q is pre-scaled by 1/sqrt(64) in its projection.
// ---------------------------------------------------------------------------
__global__ __launch_bounds__(256) void attn_kernel(
    const _Float16* __restrict__ Qm, const _Float16* __restrict__ Km,
    const _Float16* __restrict__ Vt, const int* __restrict__ mask,
    _Float16* __restrict__ Om) {
    const int lane = threadIdx.x & 31;
    const int wave = threadIdx.x >> 5;
    const int hf   = lane >> 4;
    const int ln   = lane & 15;

    const int chunksPerSeq = SEQ / 128;
    int bid = blockIdx.x;
    int c = bid % chunksPerSeq;
    int h = (bid / chunksPerSeq) % N_HEADS;
    int b = bid / (chunksPerSeq * N_HEADS);

    const int qb = c * 128 + wave * 16;
    const int q  = qb + ln;
    const size_t rowBase = (size_t)b * SEQ;
    const int hc = h * HEAD_DIM;

    const _Float16* qrow = Qm + ((rowBase + q) * D_MODEL + hc);
    v16h qf0 = load_frag_k32(qrow, hf);        // d 0..31  (pre-scaled Q)
    v16h qf1 = load_frag_k32(qrow + 32, hf);   // d 32..63

    const size_t vtBase = ((size_t)(b * N_HEADS + h) * HEAD_DIM) * SEQ;

    float mrun = -3.0e38f, lrun = 0.0f;
    v8f oacc[4] = {};

    const int smax = qb + 15;
    for (int s0 = 0; s0 <= smax; s0 += 32) {
        unsigned mvec = (unsigned)__ballot(mask[rowBase + s0 + lane] != 0);

        // ---- batch-issue all fragment loads for this chunk ----
        v16h kf[2][2];
#pragma unroll
        for (int t = 0; t < 2; ++t) {
            const _Float16* krow =
                Km + ((rowBase + s0 + t * 16 + ln) * D_MODEL + hc);
            kf[t][0] = load_frag_k32(krow, hf);
            kf[t][1] = load_frag_k32(krow + 32, hf);
        }
        v16h vf[4];
#pragma unroll
        for (int t = 0; t < 4; ++t) {
            const _Float16* vtrow =
                Vt + vtBase + (size_t)(t * 16 + ln) * SEQ + s0;
            vf[t] = load_frag_k32(vtrow, hf);
        }

        // ---- scores^T: two 16(s) x 16(q) tiles ----
        v8f sc[2];
#pragma unroll
        for (int t = 0; t < 2; ++t) {
            v8f cacc = {};
            cacc = __builtin_amdgcn_wmma_f32_16x16x32_f16(
                false, kf[t][0], false, qf0, (short)0, cacc, false, false);
            cacc = __builtin_amdgcn_wmma_f32_16x16x32_f16(
                false, kf[t][1], false, qf1, (short)0, cacc, false, false);
            sc[t] = cacc;
        }

        // ---- chunk max (+ masking only on the diagonal chunk) ----
        float sval[16];
        float cmax = -3.0e38f;
        if ((s0 + 31 <= qb) && (mvec == 0xFFFFFFFFu)) {   // wave-uniform
#pragma unroll
            for (int t = 0; t < 2; ++t)
#pragma unroll
                for (int r = 0; r < 8; ++r) {
                    float v = sc[t][r];
                    sval[t * 8 + r] = v;
                    cmax = fmaxf(cmax, v);
                }
        } else {
#pragma unroll
            for (int t = 0; t < 2; ++t)
#pragma unroll
                for (int r = 0; r < 8; ++r) {
                    int sl = t * 16 + 8 * hf + r;
                    int s = s0 + sl;
                    float v = sc[t][r];
                    bool ok = (s <= q) && (((mvec >> sl) & 1u) != 0u);
                    v = ok ? v : -1.0e9f;
                    sval[t * 8 + r] = v;
                    cmax = fmaxf(cmax, v);
                }
        }
        cmax = fmaxf(cmax, __shfl_xor(cmax, 16, 32));
        float mnew = fmaxf(mrun, cmax);
        float corr = __expf(mrun - mnew);

        // ---- exp + pack P^T B-fragment (layouts match 1:1) ----
        float psum = 0.0f;
        v16h pf;
#pragma unroll
        for (int i = 0; i < 16; ++i) {
            float p = __expf(sval[i] - mnew);
            psum += p;
            pf[i] = (_Float16)p;
        }
        psum += __shfl_xor(psum, 16, 32);
        lrun = lrun * corr + psum;
        mrun = mnew;
#pragma unroll
        for (int t = 0; t < 4; ++t)
#pragma unroll
            for (int r = 0; r < 8; ++r) oacc[t][r] *= corr;

        // ---- O^T += V^T * P^T ----
#pragma unroll
        for (int t = 0; t < 4; ++t) {
            oacc[t] = __builtin_amdgcn_wmma_f32_16x16x32_f16(
                false, vf[t], false, pf, (short)0, oacc[t], false, false);
        }
    }

    // ---- normalize + store O^T: 8 consecutive d per tile -> 16B stores ----
    float invl = (lrun > 0.0f) ? (1.0f / lrun) : 0.0f;
#pragma unroll
    for (int t = 0; t < 4; ++t) {
        h8 o;
#pragma unroll
        for (int r = 0; r < 8; ++r) o[r] = (_Float16)(oacc[t][r] * invl);
        _Float16* dst = Om + (rowBase + q) * D_MODEL + hc + t * 16 + 8 * hf;
        *reinterpret_cast<h8*>(dst) = o;
    }
}

// ---------------------------------------------------------------------------
// Host-side orchestration
// ---------------------------------------------------------------------------
extern "C" void kernel_launch(void* const* d_in, const int* in_sizes, int n_in,
                              void* d_out, int out_size, void* d_ws, size_t ws_size,
                              hipStream_t stream) {
    const float* x  = (const float*)d_in[0];
    const float* tg = (const float*)d_in[1];
    const int* mask = (const int*)d_in[2];
    const float* Wq = (const float*)d_in[3];
    const float* bq = (const float*)d_in[4];
    const float* Wk = (const float*)d_in[5];
    const float* bk = (const float*)d_in[6];
    const float* Wv = (const float*)d_in[7];
    const float* bv = (const float*)d_in[8];
    const float* Wo = (const float*)d_in[9];
    const float* bo = (const float*)d_in[10];
    const float* Wt = (const float*)d_in[11];
    const float* bt = (const float*)d_in[12];

    const size_t actB = (size_t)ROWS * D_MODEL * sizeof(_Float16);    // 8 MB
    const size_t wB   = (size_t)D_MODEL * D_MODEL * sizeof(_Float16); // 2 MB

    char* ws = (char*)d_ws;
    _Float16* x16  = (_Float16*)(ws);
    _Float16* xt16 = (_Float16*)(ws + actB);
    _Float16* Wq16 = (_Float16*)(ws + 2 * actB);
    _Float16* Wk16 = (_Float16*)(ws + 2 * actB + wB);
    _Float16* Wv16 = (_Float16*)(ws + 2 * actB + 2 * wB);
    _Float16* Wo16 = (_Float16*)(ws + 2 * actB + 3 * wB);
    _Float16* Q16  = (_Float16*)(ws + 2 * actB + 4 * wB);
    _Float16* K16  = (_Float16*)(ws + 3 * actB + 4 * wB);
    _Float16* Vt16 = (_Float16*)(ws + 4 * actB + 4 * wB);  // [B,H,64,SEQ]
    _Float16* A16  = (_Float16*)(ws + 5 * actB + 4 * wB);

    // 1) time embedding + f16 conversion
    prep_kernel<<<(ROWS * (D_MODEL / 4) + 255) / 256, 256, 0, stream>>>(
        x, tg, Wt, bt, x16, xt16, ROWS);
    const int wn = D_MODEL * D_MODEL;
    f32_to_f16_kernel<<<(wn + 255) / 256, 256, 0, stream>>>(Wq, Wq16, wn);
    f32_to_f16_kernel<<<(wn + 255) / 256, 256, 0, stream>>>(Wk, Wk16, wn);
    f32_to_f16_kernel<<<(wn + 255) / 256, 256, 0, stream>>>(Wv, Wv16, wn);
    f32_to_f16_kernel<<<(wn + 255) / 256, 256, 0, stream>>>(Wo, Wo16, wn);

    // 2) QKV projections (Q pre-scaled by 1/sqrt(64); V stored transposed)
    dim3 gemmGrid((ROWS / 256) * (D_MODEL / 64));   // 256 blocks
    gemm_bias_wmma<1><<<gemmGrid, 256, 0, stream>>>(xt16, Wq16, bq, Q16,
                                                    0.125f, ROWS, D_MODEL, D_MODEL);
    gemm_bias_wmma<1><<<gemmGrid, 256, 0, stream>>>(xt16, Wk16, bk, K16,
                                                    1.0f, ROWS, D_MODEL, D_MODEL);
    gemm_bias_wmma<2><<<gemmGrid, 256, 0, stream>>>(x16, Wv16, bv, Vt16,
                                                    1.0f, ROWS, D_MODEL, D_MODEL);

    // 3) causal flash attention
    attn_kernel<<<BATCH * N_HEADS * (SEQ / 128), 256, 0, stream>>>(
        Q16, K16, Vt16, mask, A16);

    // 4) output projection (f32 out)
    gemm_bias_wmma<0><<<gemmGrid, 256, 0, stream>>>(
        A16, Wo16, bo, d_out, 1.0f, ROWS, D_MODEL, D_MODEL);
}
